// MMDoubleStreamBlock_86741159510310
// MI455X (gfx1250) — matrix operand order repaired and checked by reading, not compiled, not run on hardware
//
#include <hip/hip_runtime.h>

typedef __attribute__((ext_vector_type(16))) __bf16 v16bf;
typedef __attribute__((ext_vector_type(8)))  float  v8f;

// ---------- helpers ----------
__device__ __forceinline__ __bf16 f2bf(float f) {
  unsigned u = __builtin_bit_cast(unsigned, f);
  unsigned r = (u + 0x7FFFu + ((u >> 16) & 1u)) >> 16;
  unsigned short s = (unsigned short)r;
  return __builtin_bit_cast(__bf16, s);
}

// stable exp-based tanh GELU (lowers to v_exp_f32, no libm tanh path)
__device__ __forceinline__ float gelu_tanh(float x) {
  float z = 0.7978845608028654f * (x + 0.044715f * x * x * x);
  float az = fabsf(z);
  float t = __expf(-2.0f * az);
  float th = (1.0f - t) / (1.0f + t);
  th = (z < 0.0f) ? -th : th;
  return 0.5f * x * (1.0f + th);
}

__device__ __forceinline__ unsigned lds_u32(const __bf16* base, int elemOff) {
  return *(const unsigned*)((const char*)base + (size_t)elemOff * 2);
}

// CDNA5 async 16-byte global->LDS copy (ASYNCcnt-tracked, bypasses VGPRs)
__device__ __forceinline__ void async_copy16(void* ldsPtr, const void* gptr) {
  unsigned lds = (unsigned)(size_t)ldsPtr;              // low 32 bits = LDS offset
  unsigned long long ga = (unsigned long long)(size_t)gptr;
  asm volatile("global_load_async_to_lds_b128 %0, %1, off" ::"v"(lds), "v"(ga)
               : "memory");
}
__device__ __forceinline__ void wait_async0() {
  asm volatile("s_wait_asynccnt 0x0" ::: "memory");
}

// A fragment: 16x32 (MxK), row = M, per ISA table:
// vgpr v (0..3): K = hlf*8 + 2v,2v+1 ; vgpr v (4..7): K = 16 + hlf*8 + 2(v-4),+1
__device__ __forceinline__ v16bf load_fragA(const __bf16* lds, int row, int stride,
                                            int kOff, int hlf) {
  union { v16bf v; unsigned u[8]; } f;
  int ro = row * stride + kOff;
#pragma unroll
  for (int r = 0; r < 8; ++r) {
    int kb = ((r >> 2) << 4) + (hlf << 3) + ((r & 3) << 1);
    f.u[r] = lds_u32(lds, ro + kb);
  }
  return f.v;
}

// B fragment: 32x16 (KxN) stored transposed in LDS as Bt[col][k]:
// same K order as A: vgpr v (0..3): K = hlf*8 + 2v,2v+1 ; vgpr v (4..7): K = 16 + hlf*8 + 2(v-4),+1
__device__ __forceinline__ v16bf load_fragB(const __bf16* lds, int col, int stride,
                                            int kOff, int hlf) {
  union { v16bf v; unsigned u[8]; } f;
  int ro = col * stride + kOff + (hlf << 3);
#pragma unroll
  for (int r = 0; r < 8; ++r) f.u[r] = lds_u32(lds, ro + ((r >> 2) << 4) + ((r & 3) << 1));
  return f.v;
}

// ---------- LayerNorm + modulation ----------
__global__ __launch_bounds__(256) void ln_mod_kernel(
    const float* __restrict__ x, const float* __restrict__ table,
    const float* __restrict__ vec, int shiftRow, int scaleRow,
    __bf16* __restrict__ out) {
  __shared__ float sA[8], sB[8];
  int rowi = blockIdx.x;
  int tid = threadIdx.x;
  const float* xr = x + (size_t)rowi * 2048;
  int d0 = tid * 8;
  float v[8]; float s = 0.f, s2 = 0.f;
#pragma unroll
  for (int j = 0; j < 8; ++j) { v[j] = xr[d0 + j]; s += v[j]; s2 += v[j] * v[j]; }
#pragma unroll
  for (int m = 16; m >= 1; m >>= 1) { s += __shfl_xor(s, m, 32); s2 += __shfl_xor(s2, m, 32); }
  int lane = tid & 31, wv = tid >> 5;
  if (lane == 0) { sA[wv] = s; sB[wv] = s2; }
  __syncthreads();
  float ts = 0.f, ts2 = 0.f;
#pragma unroll
  for (int w = 0; w < 8; ++w) { ts += sA[w]; ts2 += sB[w]; }
  float mu = ts * (1.0f / 2048.0f);
  float var = ts2 * (1.0f / 2048.0f) - mu * mu;
  float rs = rsqrtf(var + 1e-6f);
#pragma unroll
  for (int j = 0; j < 8; ++j) {
    int d = d0 + j;
    float sc = table[scaleRow * 2048 + d] + vec[d];
    float sh = table[shiftRow * 2048 + d] + vec[d];
    out[(size_t)rowi * 2048 + d] = f2bf((v[j] - mu) * rs * (1.0f + sc) + sh);
  }
}

// ---------- WMMA GEMM: double-buffered LDS, async A-tile copies ----------
// mode 0: f32 out = acc + bias
// mode 1: bf16 out = gelu(acc + bias)
// mode 2: f32 out = resid + (acc + bias) * (gate_a + gate_b)
__global__ __launch_bounds__(256) void gemm_kernel(
    const __bf16* __restrict__ A, const float* __restrict__ Bf,
    const float* __restrict__ bias,
    const float* __restrict__ gate_a, const float* __restrict__ gate_b,
    const float* __restrict__ resid, void* __restrict__ outp,
    int M, int N, int K, int mode) {
  __shared__ __align__(16) __bf16 Asm[2][128 * 40];
  __shared__ __align__(16) __bf16 Bsm[2][128 * 40];
  int tid = threadIdx.x;
  int lane = tid & 31, wv = tid >> 5;
  int hlf = lane >> 4, l16 = lane & 15;
  int wm = wv & 3, wn = wv >> 2;
  int gm0 = blockIdx.y * 128, gn0 = blockIdx.x * 128;
  (void)M;

  int arow[2], acol[2];
#pragma unroll
  for (int i = 0; i < 2; ++i) { int c = tid + i * 256; arow[i] = c >> 2; acol[i] = (c & 3) << 3; }
  int brow[4], bcol[4];
#pragma unroll
  for (int i = 0; i < 4; ++i) { int c = tid + i * 256; brow[i] = c >> 5; bcol[i] = (c & 31) << 2; }

  v8f zero = {};
  v8f acc[2][4];
#pragma unroll
  for (int i = 0; i < 2; ++i)
#pragma unroll
    for (int j = 0; j < 4; ++j) acc[i][j] = zero;

  int nk = K >> 5;
  float4 breg[4];

  // prologue: tile 0 into buffer 0
#pragma unroll
  for (int i = 0; i < 2; ++i)
    async_copy16(&Asm[0][arow[i] * 40 + acol[i]],
                 A + (size_t)(gm0 + arow[i]) * K + acol[i]);
#pragma unroll
  for (int i = 0; i < 4; ++i)
    breg[i] = *(const float4*)(Bf + (size_t)brow[i] * N + gn0 + bcol[i]);
#pragma unroll
  for (int i = 0; i < 4; ++i) {
    Bsm[0][(bcol[i] + 0) * 40 + brow[i]] = f2bf(breg[i].x);
    Bsm[0][(bcol[i] + 1) * 40 + brow[i]] = f2bf(breg[i].y);
    Bsm[0][(bcol[i] + 2) * 40 + brow[i]] = f2bf(breg[i].z);
    Bsm[0][(bcol[i] + 3) * 40 + brow[i]] = f2bf(breg[i].w);
  }
  wait_async0();
  __syncthreads();

  for (int it = 0; it < nk; ++it) {
    int cur = it & 1, nxt = cur ^ 1;
    bool more = (it + 1) < nk;
    int k1 = (it + 1) << 5;
    if (more) {  // prefetch next tile while this one computes
#pragma unroll
      for (int i = 0; i < 2; ++i)
        async_copy16(&Asm[nxt][arow[i] * 40 + acol[i]],
                     A + (size_t)(gm0 + arow[i]) * K + k1 + acol[i]);
#pragma unroll
      for (int i = 0; i < 4; ++i)
        breg[i] = *(const float4*)(Bf + (size_t)(k1 + brow[i]) * N + gn0 + bcol[i]);
    }
    v16bf afr[2], bfr[4];
#pragma unroll
    for (int mt = 0; mt < 2; ++mt)
      afr[mt] = load_fragA(Asm[cur], wm * 32 + mt * 16 + l16, 40, 0, hlf);
#pragma unroll
    for (int nt = 0; nt < 4; ++nt)
      bfr[nt] = load_fragB(Bsm[cur], wn * 64 + nt * 16 + l16, 40, 0, hlf);
#pragma unroll
    for (int mt = 0; mt < 2; ++mt)
#pragma unroll
      for (int nt = 0; nt < 4; ++nt)
        acc[mt][nt] = __builtin_amdgcn_wmma_f32_16x16x32_bf16(
            false, afr[mt], false, bfr[nt], (short)0, acc[mt][nt], false, false);
    if (more) {
#pragma unroll
      for (int i = 0; i < 4; ++i) {
        Bsm[nxt][(bcol[i] + 0) * 40 + brow[i]] = f2bf(breg[i].x);
        Bsm[nxt][(bcol[i] + 1) * 40 + brow[i]] = f2bf(breg[i].y);
        Bsm[nxt][(bcol[i] + 2) * 40 + brow[i]] = f2bf(breg[i].z);
        Bsm[nxt][(bcol[i] + 3) * 40 + brow[i]] = f2bf(breg[i].w);
      }
      wait_async0();
    }
    __syncthreads();
  }

#pragma unroll
  for (int mt = 0; mt < 2; ++mt)
#pragma unroll
    for (int nt = 0; nt < 4; ++nt)
#pragma unroll
      for (int i = 0; i < 8; ++i) {
        int m = gm0 + wm * 32 + mt * 16 + i + hlf * 8;
        int n = gn0 + wn * 64 + nt * 16 + l16;
        float v = acc[mt][nt][i] + bias[n];
        size_t idx = (size_t)m * N + n;
        if (mode == 0) {
          ((float*)outp)[idx] = v;
        } else if (mode == 1) {
          ((__bf16*)outp)[idx] = f2bf(gelu_tanh(v));
        } else {
          float g = gate_a[n] + gate_b[n];
          ((float*)outp)[idx] = resid[idx] + v * g;
        }
      }
}

// ---------- QKV pack: per-head RMSNorm + RoPE(img), write Q,K row-major & V^T ----------
__global__ __launch_bounds__(256) void qkv_pack_kernel(
    const float* __restrict__ qkv, const float* __restrict__ qnw_img,
    const float* __restrict__ knw_img, const float* __restrict__ qnw_txt,
    const float* __restrict__ knw_txt, const float* __restrict__ cs, int Li, int L,
    __bf16* __restrict__ qb, __bf16* __restrict__ kb, __bf16* __restrict__ vtb) {
  int l = blockIdx.x;
  int tid = threadIdx.x;
  int h = tid >> 4;
  int d0 = (tid & 15) << 3;
  bool isImg = (l < Li);
  const float* qn = isImg ? qnw_img : qnw_txt;
  const float* kn = isImg ? knw_img : knw_txt;
  const float* row = qkv + (size_t)l * 6144;

  for (int sel = 0; sel < 2; ++sel) {
    const float* src = row + sel * 2048 + h * 128 + d0;
    const float* w = sel ? kn : qn;
    float x[8]; float ss = 0.f;
#pragma unroll
    for (int j = 0; j < 8; ++j) { x[j] = src[j]; ss += x[j] * x[j]; }
#pragma unroll
    for (int m = 8; m >= 1; m >>= 1) ss += __shfl_xor(ss, m, 32);
    float r = rsqrtf(ss * (1.0f / 128.0f) + 1e-6f);
#pragma unroll
    for (int j = 0; j < 8; ++j) x[j] = x[j] * r * w[d0 + j];
    if (isImg) {
#pragma unroll
      for (int t = 0; t < 4; ++t) {
        int i = (d0 >> 1) + t;
        float c = cs[(size_t)l * 128 + i];
        float s = cs[(size_t)l * 128 + 64 + i];
        float x1 = x[2 * t], x2 = x[2 * t + 1];
        x[2 * t] = x1 * c - x2 * s;
        x[2 * t + 1] = x2 * c + x1 * s;
      }
    }
    __bf16* dst = (sel ? kb : qb) + ((size_t)h * L + l) * 128 + d0;
#pragma unroll
    for (int j = 0; j < 8; ++j) dst[j] = f2bf(x[j]);
  }
  const float* vsrc = row + 4096 + h * 128 + d0;
#pragma unroll
  for (int j = 0; j < 8; ++j)
    vtb[((size_t)h * 128 + d0 + j) * L + l] = f2bf(vsrc[j]);
}

// ---------- flash attention: async-double-buffered K/V, one (head, 128 q) per WG ----------
#define ATT_QS   0
#define ATT_KS0  34816
#define ATT_VS0  104448
#define ATT_BUFSTRIDE 34816
#define ATT_SF   174080
#define ATT_M    241664
#define ATT_L    242176
#define ATT_C    242688
#define ATT_SMEM 243200

__global__ __launch_bounds__(256) void attn_kernel(
    const __bf16* __restrict__ qb, const __bf16* __restrict__ kb,
    const __bf16* __restrict__ vtb, __bf16* __restrict__ attnb, int L) {
  extern __shared__ char smem[];
  __bf16* Qs = (__bf16*)(smem + ATT_QS);  // [128][136]
  float*  Sf = (float*)(smem + ATT_SF);   // [128][132] f32 scores
  __bf16* Ps = (__bf16*)(smem + ATT_SF);  // alias, row stride 264 bf16 (byte-safe)
  float* mst = (float*)(smem + ATT_M);
  float* lst = (float*)(smem + ATT_L);
  float* cst = (float*)(smem + ATT_C);

  int tid = threadIdx.x;
  int lane = tid & 31, wv = tid >> 5;
  int hlf = lane >> 4, l16 = lane & 15;
  int h = blockIdx.y;
  int qr0 = blockIdx.x * 128;
  const float scale = 0.08838834764831845f;  // 1/sqrt(128)

  int crow[8], ccol[8];
#pragma unroll
  for (int i = 0; i < 8; ++i) { int c = tid + i * 256; crow[i] = c >> 3; ccol[i] = (c & 7) << 3; }

  // prologue: async Q + K/V block 0 (buffer 0)
#pragma unroll
  for (int i = 0; i < 8; ++i)
    async_copy16(Qs + crow[i] * 136 + ccol[i],
                 qb + ((size_t)h * L + qr0 + crow[i]) * 128 + ccol[i]);
  {
    __bf16* Ks0 = (__bf16*)(smem + ATT_KS0);
    __bf16* Vs0 = (__bf16*)(smem + ATT_VS0);
#pragma unroll
    for (int i = 0; i < 8; ++i) {
      async_copy16(Ks0 + crow[i] * 136 + ccol[i],
                   kb + ((size_t)h * L + crow[i]) * 128 + ccol[i]);
      async_copy16(Vs0 + crow[i] * 136 + ccol[i],
                   vtb + ((size_t)h * 128 + crow[i]) * L + ccol[i]);
    }
  }
  if (tid < 128) { mst[tid] = -1e30f; lst[tid] = 0.0f; }

  v8f zero = {};
  v8f oacc[8];
#pragma unroll
  for (int i = 0; i < 8; ++i) oacc[i] = zero;

  wait_async0();
  __syncthreads();

  int nblk = L >> 7;
  for (int j = 0; j < nblk; ++j) {
    int cur = j & 1;
    // runtime-computed double-buffer pointers (no LDS pointer arrays)
    __bf16* KsCur = (__bf16*)(smem + ATT_KS0 + cur * ATT_BUFSTRIDE);
    __bf16* VsCur = (__bf16*)(smem + ATT_VS0 + cur * ATT_BUFSTRIDE);
    __bf16* KsNxt = (__bf16*)(smem + ATT_KS0 + (cur ^ 1) * ATT_BUFSTRIDE);
    __bf16* VsNxt = (__bf16*)(smem + ATT_VS0 + (cur ^ 1) * ATT_BUFSTRIDE);
    bool more = (j + 1) < nblk;
    if (more) {  // prefetch next K/V block into the other buffer
      int kr1 = (j + 1) << 7;
#pragma unroll
      for (int i = 0; i < 8; ++i) {
        async_copy16(KsNxt + crow[i] * 136 + ccol[i],
                     kb + ((size_t)h * L + kr1 + crow[i]) * 128 + ccol[i]);
        async_copy16(VsNxt + crow[i] * 136 + ccol[i],
                     vtb + ((size_t)h * 128 + crow[i]) * L + kr1 + ccol[i]);
      }
    }
    // S = Q K^T over this key block
    v8f sacc[8];
#pragma unroll
    for (int i = 0; i < 8; ++i) sacc[i] = zero;
#pragma unroll
    for (int ks = 0; ks < 4; ++ks) {
      v16bf a = load_fragA(Qs, wv * 16 + l16, 136, ks * 32, hlf);
#pragma unroll
      for (int nt = 0; nt < 8; ++nt) {
        v16bf b = load_fragB(KsCur, nt * 16 + l16, 136, ks * 32, hlf);
        sacc[nt] = __builtin_amdgcn_wmma_f32_16x16x32_bf16(
            false, a, false, b, (short)0, sacc[nt], false, false);
      }
    }
#pragma unroll
    for (int nt = 0; nt < 8; ++nt)
#pragma unroll
      for (int i = 0; i < 8; ++i)
        Sf[(wv * 16 + i + hlf * 8) * 132 + nt * 16 + l16] = sacc[nt][i] * scale;
    __syncthreads();

    // online softmax, one thread per query row
    if (tid < 128) {
      float mold = mst[tid];
      float mnew = mold;
      const float* srow = Sf + tid * 132;
      for (int c = 0; c < 128; ++c) mnew = fmaxf(mnew, srow[c]);
      float corr = __expf(mold - mnew);
      float rsum = 0.0f;
      __bf16* prow = Ps + (size_t)tid * 264;
      for (int c = 0; c < 128; ++c) {  // write byte 2c after read byte 4c: safe alias
        float p = __expf(srow[c] - mnew);
        rsum += p;
        prow[c] = f2bf(p);
      }
      mst[tid] = mnew;
      lst[tid] = lst[tid] * corr + rsum;
      cst[tid] = corr;
    }
    __syncthreads();

    // rescale O, accumulate P V
#pragma unroll
    for (int i = 0; i < 8; ++i) {
      float cf = cst[wv * 16 + i + hlf * 8];
#pragma unroll
      for (int nt = 0; nt < 8; ++nt) oacc[nt][i] = oacc[nt][i] * cf;
    }
#pragma unroll
    for (int ks = 0; ks < 4; ++ks) {
      v16bf a = load_fragA(Ps, wv * 16 + l16, 264, ks * 32, hlf);
#pragma unroll
      for (int nt = 0; nt < 8; ++nt) {
        v16bf b = load_fragB(VsCur, nt * 16 + l16, 136, ks * 32, hlf);
        oacc[nt] = __builtin_amdgcn_wmma_f32_16x16x32_bf16(
            false, a, false, b, (short)0, oacc[nt], false, false);
      }
    }
    if (more) wait_async0();
    __syncthreads();
  }

#pragma unroll
  for (int i = 0; i < 8; ++i) {
    int r = wv * 16 + i + hlf * 8;
    float inv = 1.0f / lst[r];
    int gl = qr0 + r;
#pragma unroll
    for (int nt = 0; nt < 8; ++nt) {
      int col = h * 128 + nt * 16 + l16;
      attnb[(size_t)gl * 2048 + col] = f2bf(oacc[nt][i] * inv);
    }
  }
}

// ---------- orchestration ----------
extern "C" void kernel_launch(void* const* d_in, const int* in_sizes, int n_in,
                              void* d_out, int out_size, void* d_ws, size_t ws_size,
                              hipStream_t stream) {
  (void)in_sizes; (void)n_in; (void)out_size; (void)ws_size;
  const int Li = 2048, Lt = 256, L = 2304, D = 2048, Dff = 8192, H = 16;

  const float* img    = (const float*)d_in[0];
  const float* txt    = (const float*)d_in[1];
  const float* vec    = (const float*)d_in[2];
  const float* cs     = (const float*)d_in[3];
  const float* imod   = (const float*)d_in[4];
  const float* iqkvw  = (const float*)d_in[5];
  const float* iqkvb  = (const float*)d_in[6];
  const float* iqn    = (const float*)d_in[7];
  const float* ikn    = (const float*)d_in[8];
  const float* iprojw = (const float*)d_in[9];
  const float* iprojb = (const float*)d_in[10];
  const float* ifc1w  = (const float*)d_in[11];
  const float* ifc1b  = (const float*)d_in[12];
  const float* ifc2w  = (const float*)d_in[13];
  const float* ifc2b  = (const float*)d_in[14];
  const float* tmod   = (const float*)d_in[15];
  const float* tqkvw  = (const float*)d_in[16];
  const float* tqkvb  = (const float*)d_in[17];
  const float* tqn    = (const float*)d_in[18];
  const float* tkn    = (const float*)d_in[19];
  const float* tprojw = (const float*)d_in[20];
  const float* tprojb = (const float*)d_in[21];
  const float* tfc1w  = (const float*)d_in[22];
  const float* tfc1b  = (const float*)d_in[23];
  const float* tfc2w  = (const float*)d_in[24];
  const float* tfc2b  = (const float*)d_in[25];

  char* ws = (char*)d_ws;
  size_t off = 0;
  auto alloc = [&](size_t bytes) {
    void* p = ws + off;
    off += (bytes + 255) & ~(size_t)255;
    return p;
  };
  __bf16* ln1   = (__bf16*)alloc((size_t)L * D * 2);
  float*  qkv   = (float*) alloc((size_t)L * 3 * D * 4);
  __bf16* qb    = (__bf16*)alloc((size_t)H * L * 128 * 2);
  __bf16* kb    = (__bf16*)alloc((size_t)H * L * 128 * 2);
  __bf16* vtb   = (__bf16*)alloc((size_t)H * L * 128 * 2);
  __bf16* attnb = (__bf16*)alloc((size_t)L * D * 2);
  float*  mid   = (float*) alloc((size_t)L * D * 4);
  __bf16* ln2   = (__bf16*)alloc((size_t)L * D * 2);
  __bf16* hbuf  = (__bf16*)alloc((size_t)L * Dff * 2);

  float* outImg = (float*)d_out;
  float* outTxt = outImg + (size_t)Li * D;

  // 1. LN1 + modulation (rows 0=shift1, 1=scale1)
  ln_mod_kernel<<<Li, 256, 0, stream>>>(img, imod, vec, 0, 1, ln1);
  ln_mod_kernel<<<Lt, 256, 0, stream>>>(txt, tmod, vec, 0, 1, ln1 + (size_t)Li * D);
  // 2. QKV GEMMs
  gemm_kernel<<<dim3(3 * D / 128, Li / 128), 256, 0, stream>>>(
      ln1, iqkvw, iqkvb, nullptr, nullptr, nullptr, qkv, Li, 3 * D, D, 0);
  gemm_kernel<<<dim3(3 * D / 128, Lt / 128), 256, 0, stream>>>(
      ln1 + (size_t)Li * D, tqkvw, tqkvb, nullptr, nullptr, nullptr,
      qkv + (size_t)Li * 3 * D, Lt, 3 * D, D, 0);
  // 3. RMSNorm + RoPE + pack
  qkv_pack_kernel<<<L, 256, 0, stream>>>(qkv, iqn, ikn, tqn, tkn, cs, Li, L, qb, kb, vtb);
  // 4. flash attention
  attn_kernel<<<dim3(L / 128, H), 256, ATT_SMEM, stream>>>(qb, kb, vtb, attnb, L);
  // 5. proj + gate1 + residual
  gemm_kernel<<<dim3(D / 128, Li / 128), 256, 0, stream>>>(
      attnb, iprojw, iprojb, imod + 2 * D, vec, img, mid, Li, D, D, 2);
  gemm_kernel<<<dim3(D / 128, Lt / 128), 256, 0, stream>>>(
      attnb + (size_t)Li * D, tprojw, tprojb, tmod + 2 * D, vec, txt,
      mid + (size_t)Li * D, Lt, D, D, 2);
  // 6. LN2 + modulation (rows 3=shift2, 4=scale2)
  ln_mod_kernel<<<Li, 256, 0, stream>>>(mid, imod, vec, 3, 4, ln2);
  ln_mod_kernel<<<Lt, 256, 0, stream>>>(mid + (size_t)Li * D, tmod, vec, 3, 4,
                                        ln2 + (size_t)Li * D);
  // 7. FC1 + GELU
  gemm_kernel<<<dim3(Dff / 128, Li / 128), 256, 0, stream>>>(
      ln2, ifc1w, ifc1b, nullptr, nullptr, nullptr, hbuf, Li, Dff, D, 1);
  gemm_kernel<<<dim3(Dff / 128, Lt / 128), 256, 0, stream>>>(
      ln2 + (size_t)Li * D, tfc1w, tfc1b, nullptr, nullptr, nullptr,
      hbuf + (size_t)Li * Dff, Lt, Dff, D, 1);
  // 8. FC2 + gate2 + residual -> outputs
  gemm_kernel<<<dim3(D / 128, Li / 128), 256, 0, stream>>>(
      hbuf, ifc2w, ifc2b, imod + 5 * D, vec, mid, outImg, Li, D, Dff, 2);
  gemm_kernel<<<dim3(D / 128, Lt / 128), 256, 0, stream>>>(
      hbuf + (size_t)Li * Dff, tfc2w, tfc2b, tmod + 5 * D, vec,
      mid + (size_t)Li * D, outTxt, Lt, D, Dff, 2);
}
